// ModulatedConv2d_30494267801815
// MI455X (gfx1250) — matrix operand order repaired and checked by reading, not compiled, not run on hardware
//
#include <hip/hip_runtime.h>
#include <hip/hip_bf16.h>
#include <math.h>

typedef __bf16 bf16_t;
typedef __attribute__((ext_vector_type(8)))  bf16_t v8bf;
typedef __attribute__((ext_vector_type(16))) bf16_t v16bf;
typedef __attribute__((ext_vector_type(8)))  float  v8f;

static constexpr int NB   = 16;    // batch
static constexpr int CIN  = 512;
static constexpr int COUT = 512;
static constexpr int IMH  = 64;
static constexpr int IMW  = 64;
static constexpr int HW   = IMH * IMW;   // 4096
static constexpr int KTAP = 9;           // 3x3
static constexpr int KTOT = CIN * KTAP;  // 4608

static constexpr int PIXT  = 256;        // pixels per block tile (4 image rows)
static constexpr int SLABR = 6;          // 4 rows + halo
static constexpr int SLABC = 66;         // 64 cols + halo
static constexpr int CELLS = SLABR * SLABC;  // 396
static constexpr int CSTR  = 40;         // bf16 per slab cell (32 + pad), 80B

// ---------------------------------------------------------------------------
// K1: s[b, cin] = style[b, :] @ mod_w[cin, :] + mod_b[cin]
// ---------------------------------------------------------------------------
__global__ void style_mod_kernel(const float* __restrict__ style,
                                 const float* __restrict__ mod_w,
                                 const float* __restrict__ mod_b,
                                 float* __restrict__ s_out) {
    const int b   = blockIdx.x;
    const int cin = threadIdx.x;
    const float* st = style + b * CIN;
    const float* mw = mod_w + (size_t)cin * CIN;
    float acc = mod_b[cin];
    for (int k = 0; k < CIN; ++k) acc = fmaf(st[k], mw[k], acc);
    s_out[b * CIN + cin] = acc;
}

// ---------------------------------------------------------------------------
// K2: per-(b,cout) modulate + demodulate, write bf16 weights laid out
//     wmod[((b*9 + tap)*COUT + cout)*CIN + cin]   (K = cin contiguous)
// ---------------------------------------------------------------------------
__global__ void modulate_kernel(const float* __restrict__ weight,
                                const float* __restrict__ s,
                                bf16_t* __restrict__ wmod) {
    const int blk  = blockIdx.x;
    const int b    = blk >> 9;
    const int cout = blk & (COUT - 1);
    const int t    = threadIdx.x;
    const float scale = 0.0147313913f;   // 1/sqrt(512*9)

    const float* wrow = weight + (size_t)cout * KTOT;
    const float* srow = s + b * CIN;

    float vals[KTOT / 256];              // 18 per thread
    float sq = 0.f;
#pragma unroll
    for (int j = 0; j < KTOT / 256; ++j) {
        const int e   = t + j * 256;
        const int cin = e / 9;
        const float v = scale * wrow[e] * srow[cin];
        vals[j] = v;
        sq += v * v;
    }
    __shared__ float red[256];
    red[t] = sq;
    __syncthreads();
    for (int stp = 128; stp > 0; stp >>= 1) {
        if (t < stp) red[t] += red[t + stp];
        __syncthreads();
    }
    const float demod = rsqrtf(red[0] + 1e-8f);
#pragma unroll
    for (int j = 0; j < KTOT / 256; ++j) {
        const int e   = t + j * 256;
        const int cin = e / 9;
        const int tap = e - cin * 9;
        wmod[(((size_t)b * KTAP + tap) * COUT + cout) * CIN + cin] =
            (bf16_t)(vals[j] * demod);
    }
}

// ---------------------------------------------------------------------------
// K3: implicit-GEMM conv, bf16 WMMA, LDS halo-slab for all 9 taps.
//   grid = (16 pixel-tiles, 8 cout-tiles, 16 batch), block = 256 (8 waves)
//   block tile: 64 cout x 256 pixels; wave tile: 32 x 64 (2x4 WMMA C-tiles)
// ---------------------------------------------------------------------------
__global__ __launch_bounds__(256) void conv_wmma_kernel(
        const float* __restrict__ x,
        const bf16_t* __restrict__ wmod,
        const float* __restrict__ bias,
        float* __restrict__ out) {
    const int tid  = threadIdx.x;
    const int lane = tid & 31;
    const int wave = tid >> 5;
    const int wm   = wave & 1;   // cout dim (2 waves of 32)
    const int wn   = wave >> 1;  // pixel dim (4 waves of 64)

    const int b     = blockIdx.z;
    const int cout0 = blockIdx.y * 64;
    const int n0    = blockIdx.x * PIXT;  // 4 full image rows
    const int h0    = n0 >> 6;

    // x halo slab: [6 rows][66 cols][32 cin], cell stride 40 bf16 (80B)
    __shared__ bf16_t Xs[CELLS * CSTR];           // 31,680 B
    __shared__ bf16_t As[2][64 * 40];             // 2 x 5,120 B (ping-pong)

    v8f acc[2][4] = {};

    const float* xb = x + (size_t)b * CIN * HW;

    // A loader mapping: one uint4 (8 bf16) per thread per tap
    const int arow = tid >> 2;          // 0..63
    const int akb  = (tid & 3) * 8;     // 0/8/16/24

    // per-lane fragment constants
    const int l15  = lane & 15;
    const int kbA  = (lane >> 4) * 8;   // A: lanes<16 K0-7,16-23 ; else 8-15,24-31
    const int kbB  = (lane >> 4) * 8;   // B: lanes<16 K0-7,16-23 ; else 8-15,24-31

    const bf16_t* awb = wmod + (((size_t)b * KTAP) * COUT + cout0) * CIN;

    for (int cin0 = 0; cin0 < CIN; cin0 += 32) {
        // ---- stage x halo slab (fp32 -> bf16), zero the padding halo ----
        for (int cell = tid; cell < CELLS; cell += 256) {
            const int r  = cell / SLABC;
            const int c  = cell - r * SLABC;
            const int gr = h0 - 1 + r;
            const int gc = c - 1;
            const bool inb = ((unsigned)gr < (unsigned)IMH) &&
                             ((unsigned)gc < (unsigned)IMW);
            const float* xp = xb + (size_t)cin0 * HW + (gr * IMW + gc);
            v8bf t[4];
#pragma unroll
            for (int i = 0; i < 32; ++i) {
                const float v = inb ? xp[(size_t)i * HW] : 0.f;
                t[i >> 3][i & 7] = (bf16_t)v;
            }
#pragma unroll
            for (int q = 0; q < 4; ++q)
                *reinterpret_cast<v8bf*>(&Xs[cell * CSTR + q * 8]) = t[q];
            // pull next cin-chunk's lines toward L1/L2 (global_prefetch_b8)
            if (inb && (cin0 + 32 < CIN))
                __builtin_prefetch(xp + (size_t)32 * HW, 0, 3);
        }
        // ---- A tile for tap 0 into buffer 0 ----
        {
            const uint4 v = *reinterpret_cast<const uint4*>(
                awb + (size_t)arow * CIN + cin0 + akb);
            *reinterpret_cast<uint4*>(&As[0][arow * 40 + akb]) = v;
        }
        __syncthreads();

        for (int tap = 0; tap < KTAP; ++tap) {
            const int buf = tap & 1;
            // preload next tap's A tile into registers (hidden under WMMAs)
            uint4 nxt;
            if (tap < KTAP - 1) {
                nxt = *reinterpret_cast<const uint4*>(
                    awb + ((size_t)(tap + 1) * COUT) * CIN +
                    (size_t)arow * CIN + cin0 + akb);
            }
            const int dh = tap / 3 - 1;
            const int dw = tap % 3 - 1;

            // A fragments (rows wm*32 + {0,16} + lane%16)
            v16bf afrag[2];
#pragma unroll
            for (int i = 0; i < 2; ++i) {
                const int row = wm * 32 + i * 16 + l15;
                v8bf lo = *reinterpret_cast<const v8bf*>(&As[buf][row * 40 + kbA]);
                v8bf hi = *reinterpret_cast<const v8bf*>(&As[buf][row * 40 + kbA + 16]);
                afrag[i] = __builtin_shufflevector(lo, hi,
                    0,1,2,3,4,5,6,7,8,9,10,11,12,13,14,15);
            }
            // 4 B fragments from the slab, 2 WMMAs each
#pragma unroll
            for (int j = 0; j < 4; ++j) {
                const int p    = wn * 64 + j * 16 + l15;
                const int cell = ((p >> 6) + dh + 1) * SLABC + ((p & 63) + dw + 1);
                v8bf lo = *reinterpret_cast<const v8bf*>(&Xs[cell * CSTR + kbB]);
                v8bf hi = *reinterpret_cast<const v8bf*>(&Xs[cell * CSTR + kbB + 16]);
                v16bf bfrag = __builtin_shufflevector(lo, hi,
                    0,1,2,3,4,5,6,7,8,9,10,11,12,13,14,15);
                acc[0][j] = __builtin_amdgcn_wmma_f32_16x16x32_bf16(
                    false, afrag[0], false, bfrag, (short)0, acc[0][j], false, false);
                acc[1][j] = __builtin_amdgcn_wmma_f32_16x16x32_bf16(
                    false, afrag[1], false, bfrag, (short)0, acc[1][j], false, false);
            }
            // commit next A tile to the other buffer
            if (tap < KTAP - 1)
                *reinterpret_cast<uint4*>(&As[buf ^ 1][arow * 40 + akb]) = nxt;
            __syncthreads();
        }
    }

    // ---- store + bias; C layout: M = r + 8*(lane/16), N = lane%16 ----
#pragma unroll
    for (int i = 0; i < 2; ++i) {
#pragma unroll
        for (int r = 0; r < 8; ++r) {
            const int m    = r + 8 * (lane >> 4);
            const int cout = cout0 + wm * 32 + i * 16 + m;
            const float bv = bias[cout];
            float* orow = out + ((size_t)b * COUT + cout) * HW + n0;
#pragma unroll
            for (int j = 0; j < 4; ++j)
                orow[wn * 64 + j * 16 + l15] = acc[i][j][r] + bv;
        }
    }
}

// ---------------------------------------------------------------------------
extern "C" void kernel_launch(void* const* d_in, const int* in_sizes, int n_in,
                              void* d_out, int out_size, void* d_ws, size_t ws_size,
                              hipStream_t stream) {
    const float* x      = (const float*)d_in[0];  // [16,512,64,64]
    const float* style  = (const float*)d_in[1];  // [16,512]
    const float* weight = (const float*)d_in[2];  // [512,512,3,3]
    const float* bias   = (const float*)d_in[3];  // [1,512,1,1]
    const float* mod_w  = (const float*)d_in[4];  // [512,512]
    const float* mod_b  = (const float*)d_in[5];  // [512]
    float* out = (float*)d_out;

    // workspace: s (32KB) | wmod bf16 (75.5MB), 64KB-aligned split
    float*  s_ws = (float*)d_ws;
    bf16_t* wmod = (bf16_t*)((char*)d_ws + (64 * 1024));

    style_mod_kernel<<<dim3(NB), dim3(CIN), 0, stream>>>(style, mod_w, mod_b, s_ws);
    modulate_kernel<<<dim3(NB * COUT), dim3(256), 0, stream>>>(weight, s_ws, wmod);
    conv_wmma_kernel<<<dim3(HW / PIXT, COUT / 64, NB), dim3(256), 0, stream>>>(
        x, wmod, bias, out);
}
